// Attention_13082470384012
// MI455X (gfx1250) — matrix operand order repaired and checked by reading, not compiled, not run on hardware
//
#include <hip/hip_runtime.h>
#include <hip/hip_bf16.h>

typedef __attribute__((ext_vector_type(16))) __bf16 v16bf;
typedef __attribute__((ext_vector_type(8)))  float  v8f;

#define WMMA_BF16(a, b, c) \
  __builtin_amdgcn_wmma_f32_16x16x32_bf16(false, (a), false, (b), (short)0, (c), false, false)

// CDNA5 async global->LDS copy (ASYNCcnt-tracked), GV addressing mode.
__device__ __forceinline__ void async_load_b128(unsigned lds_off, const void* gaddr) {
  asm volatile("global_load_async_to_lds_b128 %0, %1, off"
               :: "v"(lds_off), "v"(gaddr)
               : "memory");
}
// Async loads retire in order: <=8 outstanding means everything older than the
// most recent 8 issues has landed in LDS.
__device__ __forceinline__ void wait_asynccnt_le8() {
  asm volatile("s_wait_asynccnt 0x8" ::: "memory");
}
__device__ __forceinline__ void wait_asynccnt_le0() {
  asm volatile("s_wait_asynccnt 0x0" ::: "memory");
}

// ---------------------------------------------------------------------------
// GEMM:  Y[m, n] = (X[m, :] . W[n, :]) + bias[n]   (X: 8192x512, W: 512x512)
// MODE 0: fp32 flat [8192, 512]
// MODE 1: bf16 head-split [B=2, H=8, S=4096, D=64], scaled
// MODE 2: bf16 head-split TRANSPOSED [B=2, H=8, D=64, S=4096]  (for V)
// Block: 256 threads (8 waves). Tile: 128(M) x 64(N). K-step 32.
// ---------------------------------------------------------------------------
template <bool XBF16, int MODE>
__global__ __launch_bounds__(256) void gemm512(const void* __restrict__ Xv,
                                               const float* __restrict__ W,
                                               const float* __restrict__ bias,
                                               void* __restrict__ Yv,
                                               float scale) {
  constexpr int KDIM = 512;
  __shared__ __bf16 Xl[128][40];   // row stride 80B (16B aligned, bank-spread)
  __shared__ __bf16 Wl[64][40];    // Wl[n][kk] = W[n0+n][k0+kk]

  const int tid  = threadIdx.x;
  const int lane = tid & 31;
  const int wave = tid >> 5;
  const int ncol = lane & 15;
  const int hi   = lane >> 4;

  const int m0 = (blockIdx.x >> 3) * 128;
  const int n0 = (blockIdx.x & 7) * 64;

  v8f acc[4] = {};

  const int xrow = tid >> 1;           // 0..127
  const int xcs  = (tid & 1) * 16;     // 16 elems each
  const int wrow = tid >> 2;           // 0..63
  const int wcs  = (tid & 3) * 8;      // 8 elems each

  for (int kb = 0; kb < 16; ++kb) {
    const int k0 = kb * 32;
    __syncthreads();
    if (XBF16) {
      const __bf16* src = (const __bf16*)Xv + (size_t)(m0 + xrow) * KDIM + k0 + xcs;
#pragma unroll
      for (int j = 0; j < 16; ++j) Xl[xrow][xcs + j] = src[j];
    } else {
      const float* src = (const float*)Xv + (size_t)(m0 + xrow) * KDIM + k0 + xcs;
#pragma unroll
      for (int j = 0; j < 16; ++j) Xl[xrow][xcs + j] = (__bf16)src[j];
    }
    {
      const float* wsrc = W + (size_t)(n0 + wrow) * KDIM + k0 + wcs;
#pragma unroll
      for (int j = 0; j < 8; ++j) Wl[wrow][wcs + j] = (__bf16)wsrc[j];
    }
    __syncthreads();

    // A fragment: 16(m) x 32(k). Lane m = lane&15; e<8 -> K=e+8hi, e>=8 -> K=e+8+8hi.
    v16bf a;
    {
      const __bf16* xr = &Xl[wave * 16 + ncol][0];
      uint4* ap = (uint4*)&a;
      ap[0] = *(const uint4*)(xr + 8 * hi);
      ap[1] = *(const uint4*)(xr + 16 + 8 * hi);
    }
#pragma unroll
    for (int ns = 0; ns < 4; ++ns) {
      // B fragment: 32(k) x 16(n). Lane n = lane&15; e<8 -> K=e+8hi, e>=8 -> K=e+8+8hi.
      v16bf bm;
      const __bf16* wr = &Wl[ns * 16 + ncol][8 * hi];
      uint4* bp = (uint4*)&bm;
      bp[0] = ((const uint4*)wr)[0];
      bp[1] = ((const uint4*)wr)[2];
      acc[ns] = WMMA_BF16(a, bm, acc[ns]);
    }
  }

  // Epilogue. C layout: lane col = ncol, VGPR r -> row r + 8*hi.
#pragma unroll
  for (int ns = 0; ns < 4; ++ns) {
    const int c  = n0 + ns * 16 + ncol;
    const float bb = bias[c];
#pragma unroll
    for (int r = 0; r < 8; ++r) {
      const int m  = m0 + wave * 16 + r + 8 * hi;
      const float v = (acc[ns][r] + bb) * scale;
      const int b = m >> 12, s = m & 4095;
      const int h = c >> 6,  d = c & 63;
      if (MODE == 1) {
        ((__bf16*)Yv)[(((size_t)(b * 8 + h)) * 4096 + s) * 64 + d] = (__bf16)v;
      } else if (MODE == 2) {
        ((__bf16*)Yv)[(((size_t)(b * 8 + h)) * 64 + d) * 4096 + s] = (__bf16)v;
      } else {
        ((float*)Yv)[(size_t)m * 512 + c] = v;
      }
    }
  }
}

// ---------------------------------------------------------------------------
// Flash attention. Block: 128 threads (4 waves). One (b,h), 64 query rows.
// Q/K: bf16 [B,H,S,64] (Q pre-scaled by 1/sqrt(D)). Vt: bf16 [B,H,64,S].
// Ctx out: bf16 [B,S,512].
// Double-buffered ASYNC global->LDS pipeline: tile kt+1 streams in while
// WMMAs run on tile kt.
// ---------------------------------------------------------------------------
__global__ __launch_bounds__(128) void flash_attn(const __bf16* __restrict__ Q,
                                                  const __bf16* __restrict__ K,
                                                  const __bf16* __restrict__ Vt,
                                                  __bf16* __restrict__ Ctx) {
  __shared__ __bf16 Kl[2][64][72];    // [buf][k_local][d]
  __shared__ __bf16 Vl[2][64][72];    // [buf][d][k_local]  (V pre-transposed)
  __shared__ __bf16 Pl[4][16][72];    // per-wave P staging [q_local][k_local]

  const int tid  = threadIdx.x;
  const int lane = tid & 31;
  const int wave = tid >> 5;
  const int ncol = lane & 15;
  const int hi   = lane >> 4;

  const int qt = blockIdx.x & 63;     // q-tile of 64
  const int bh = blockIdx.x >> 6;     // b*8 + h
  const size_t base = (size_t)bh * 4096 * 64;
  const int q0 = qt * 64 + wave * 16;

  const int row = tid >> 1;           // 0..63 (k row for K tile, d row for V tile)
  const int cs  = (tid & 1) * 32;     // 32 elems each

  // Per-thread source pointers for tile kt.
  const __bf16* kg0 = K  + base + (size_t)row * 64 + cs;    // + kt*4096
  const __bf16* vg0 = Vt + base + (size_t)row * 4096 + cs;  // + kt*64

  // Issue 8 async B128 copies for tile kt into buffer buf.
  auto issue_tile = [&](int kt, int buf) {
    const __bf16* kg = kg0 + (size_t)kt * 64 * 64;
    const __bf16* vg = vg0 + (size_t)kt * 64;
#pragma unroll
    for (int j = 0; j < 32; j += 8) {
      async_load_b128((unsigned)(size_t)&Kl[buf][row][cs + j], kg + j);
      async_load_b128((unsigned)(size_t)&Vl[buf][row][cs + j], vg + j);
    }
  };

  // Preload this wave's Q A-fragments (16 rows x 64 d => 2 K-steps of 32).
  v16bf qa[2];
  {
    const __bf16* qr = Q + base + (size_t)(q0 + ncol) * 64;
#pragma unroll
    for (int ks = 0; ks < 2; ++ks) {
      uint4* ap = (uint4*)&qa[ks];
      ap[0] = *(const uint4*)(qr + 32 * ks + 8 * hi);
      ap[1] = *(const uint4*)(qr + 32 * ks + 16 + 8 * hi);
    }
  }

  float mrun[8], lrun[8];
#pragma unroll
  for (int r = 0; r < 8; ++r) { mrun[r] = -3.0e38f; lrun[r] = 0.0f; }
  v8f O[4] = {};

  issue_tile(0, 0);  // prime the pipeline

  for (int kt = 0; kt < 64; ++kt) {
    const int buf = kt & 1;

    // All waves finished computing on buf^1 (used last iteration), so it
    // is safe to start overwriting it with tile kt+1.
    __syncthreads();
    if (kt + 1 < 64) {
      issue_tile(kt + 1, buf ^ 1);
      wait_asynccnt_le8();   // tile kt landed; tile kt+1 still in flight
    } else {
      wait_asynccnt_le0();
    }
    __syncthreads();         // tile kt visible to every wave

    // S = Q . K^T  (16 x 64 per wave, pre-scaled)
    v8f S[4] = {};
#pragma unroll
    for (int ns = 0; ns < 4; ++ns) {
#pragma unroll
      for (int ks = 0; ks < 2; ++ks) {
        v16bf bm;
        const __bf16* kr = &Kl[buf][ns * 16 + ncol][32 * ks + 8 * hi];
        uint4* bp = (uint4*)&bm;
        bp[0] = ((const uint4*)kr)[0];
        bp[1] = ((const uint4*)kr)[2];
        S[ns] = WMMA_BF16(qa[ks], bm, S[ns]);
      }
    }

    // Online softmax. Row r+8hi lives in VGPR r across the 16-lane half.
    float mnew[8], corr[8], tsum[8];
#pragma unroll
    for (int r = 0; r < 8; ++r) {
      float v = fmaxf(fmaxf(S[0][r], S[1][r]), fmaxf(S[2][r], S[3][r]));
#pragma unroll
      for (int xm = 1; xm < 16; xm <<= 1) v = fmaxf(v, __shfl_xor(v, xm, 32));
      mnew[r] = fmaxf(mrun[r], v);
      corr[r] = __expf(mrun[r] - mnew[r]);
      mrun[r] = mnew[r];
      tsum[r] = 0.0f;
    }
#pragma unroll
    for (int ns = 0; ns < 4; ++ns) {
#pragma unroll
      for (int r = 0; r < 8; ++r) {
        const float p = __expf(S[ns][r] - mnew[r]);
        tsum[r] += p;
        Pl[wave][r + 8 * hi][ns * 16 + ncol] = (__bf16)p;
      }
    }
#pragma unroll
    for (int r = 0; r < 8; ++r) {
      float v = tsum[r];
#pragma unroll
      for (int xm = 1; xm < 16; xm <<= 1) v += __shfl_xor(v, xm, 32);
      lrun[r] = lrun[r] * corr[r] + v;
    }
#pragma unroll
    for (int ns = 0; ns < 4; ++ns)
#pragma unroll
      for (int r = 0; r < 8; ++r) O[ns][r] *= corr[r];

    // O += P . V  (per-wave Pl: DS ops are in-order within a wave)
#pragma unroll
    for (int ks = 0; ks < 2; ++ks) {
      v16bf pa;
      const __bf16* pr = &Pl[wave][ncol][0];
      uint4* ap = (uint4*)&pa;
      ap[0] = *(const uint4*)(pr + 32 * ks + 8 * hi);
      ap[1] = *(const uint4*)(pr + 32 * ks + 16 + 8 * hi);
#pragma unroll
      for (int ds = 0; ds < 4; ++ds) {
        v16bf bm;
        const __bf16* vr = &Vl[buf][ds * 16 + ncol][32 * ks + 8 * hi];
        uint4* bp = (uint4*)&bm;
        bp[0] = ((const uint4*)vr)[0];
        bp[1] = ((const uint4*)vr)[2];
        O[ds] = WMMA_BF16(pa, bm, O[ds]);
      }
    }
  }

  // Normalize and write ctx[b, q, h*64 + d] as bf16.
  const int b = bh >> 3, h = bh & 7;
#pragma unroll
  for (int ds = 0; ds < 4; ++ds) {
#pragma unroll
    for (int r = 0; r < 8; ++r) {
      const int q = q0 + r + 8 * hi;
      const float v = O[ds][r] / lrun[r];
      Ctx[((size_t)(b * 4096 + q)) * 512 + h * 64 + ds * 16 + ncol] = (__bf16)v;
    }
  }
}

// ---------------------------------------------------------------------------
extern "C" void kernel_launch(void* const* d_in, const int* in_sizes, int n_in,
                              void* d_out, int out_size, void* d_ws, size_t ws_size,
                              hipStream_t stream) {
  const float* query   = (const float*)d_in[0];
  const float* context = (const float*)d_in[1];
  const float* Wq = (const float*)d_in[2];
  const float* bq = (const float*)d_in[3];
  const float* Wk = (const float*)d_in[4];
  const float* bk = (const float*)d_in[5];
  const float* Wv = (const float*)d_in[6];
  const float* bv = (const float*)d_in[7];
  const float* Wo = (const float*)d_in[8];
  const float* bo = (const float*)d_in[9];

  const size_t elems = (size_t)2 * 8 * 4096 * 64;  // 4,194,304 per tensor
  __bf16* Qb  = (__bf16*)d_ws;
  __bf16* Kb  = Qb + elems;
  __bf16* Vtb = Kb + elems;   // V transposed: [B,H,D,S]
  __bf16* Cb  = Vtb + elems;  // ctx, bf16 [2,4096,512]; total ws use = 32 MB

  dim3 ggrid(512), gblk(256);
  // Fold 1/sqrt(HEAD_DIM)=1/8 into Q (after bias, matching reference).
  gemm512<false, 1><<<ggrid, gblk, 0, stream>>>(query,   Wq, bq, Qb,  0.125f);
  gemm512<false, 1><<<ggrid, gblk, 0, stream>>>(context, Wk, bk, Kb,  1.0f);
  gemm512<false, 2><<<ggrid, gblk, 0, stream>>>(context, Wv, bv, Vtb, 1.0f);

  flash_attn<<<dim3(16 * 64), dim3(128), 0, stream>>>(Qb, Kb, Vtb, Cb);

  gemm512<true, 0><<<ggrid, gblk, 0, stream>>>(Cb, Wo, bo, d_out, 1.0f);
}
